// PFNet7_16887811407984
// MI455X (gfx1250) — matrix operand order repaired and checked by reading, not compiled, not run on hardware
//
#include <hip/hip_runtime.h>
#include <hip/hip_bf16.h>

// ---------------------------------------------------------------------------
// PFNet7 inference on MI455X (gfx1250, wave32, WMMA + async global->LDS)
// ---------------------------------------------------------------------------

#define NNODES 12000
#define KPAD   8           // LDS row padding (u16 elements) = 16 bytes

typedef unsigned short u16;
typedef __attribute__((ext_vector_type(16))) u16    v16u;
typedef __attribute__((ext_vector_type(8)))  u16    v8u;
typedef __attribute__((ext_vector_type(8)))  float  v8f;
typedef __attribute__((ext_vector_type(16))) __bf16 v16bf;

__device__ __forceinline__ u16 f2bf(float f) {
    unsigned int u = __builtin_bit_cast(unsigned int, f);
    unsigned int r = u + 0x7FFFu + ((u >> 16) & 1u);
    return (u16)(r >> 16);
}

// ---------------------------------------------------------------------------
// Weight packing: W[K,O] f32 (row-major, [in,out]) -> bf16 B-fragments.
// Fragment (kt, ot): 32 lanes x 16 bf16 contiguous. Lane L, elem e holds
// W[kt*32 + 16*(L/16) + e][ot*16 + L%16]  (zero padded OOB).
// Contiguous k order (VGPR v = K {16h+2v, 16h+2v+1}), same as the A fragment read.
// ---------------------------------------------------------------------------
__global__ __launch_bounds__(32) void pack_w_k(const float* __restrict__ W,
                                               u16* __restrict__ out,
                                               int K, int O, int Op) {
    const int otiles = Op >> 4;
    const int frag = blockIdx.x;
    const int kt = frag / otiles, ot = frag % otiles;
    const int L = threadIdx.x;
    const int h = L >> 4, lm = L & 15;
    const int n = ot * 16 + lm;
    u16* dst = out + (size_t)frag * 512 + (size_t)L * 16;
#pragma unroll
    for (int e = 0; e < 16; ++e) {
        const int k = kt * 32 + 16 * h + e;
        float v = (k < K && n < O) ? W[(size_t)k * O + n] : 0.0f;
        dst[e] = f2bf(v);
    }
}

// ---------------------------------------------------------------------------
// WMMA GEMM: C[M=12000, O] = act(A[M,Kp](bf16, ldA==Kp) @ Wpacked + bias)
// Block = (Op/16)/OT waves, all sharing one 16-row A tile staged in LDS via
// global_load_async_to_lds_b128 (ASYNCcnt). Each wave emits a 16 x (16*OT)
// strip: one A fragment (2x ds_load_b128) feeds OT WMMAs.
// ACT: 0=none, 1=ELU, 2=LeakyReLU.  Output bf16 (outB) or f32 (outF).
// ---------------------------------------------------------------------------
template <int ACT, int OT>
__global__ void gemm_wmma_k(const u16* __restrict__ A, const u16* __restrict__ Wp,
                            const float* __restrict__ bias,
                            u16* __restrict__ outB, float* __restrict__ outF,
                            int Kp, int O, int Op, int ldOut) {
    __shared__ u16 smem[16 * (256 + KPAD)];
    const int mt = blockIdx.x;
    const int tid = (int)threadIdx.x;
    const int wave = tid >> 5, L = tid & 31;
    const int h = L >> 4, lm = L & 15;
    const int otiles = Op >> 4;
    const int ldS = Kp + KPAD;             // LDS row stride (u16 elements)

#if defined(__HIP_DEVICE_COMPILE__)
    {
        // Stage contiguous 16 x Kp bf16 tile -> padded LDS rows (async engine)
        const char* gbase = (const char*)(A + (size_t)mt * 16 * Kp);
        const unsigned lbase = (unsigned)(uintptr_t)(&smem[0]);
        const int cpr = (Kp * 2) >> 4;     // 16B chunks per row
        for (int c = tid; c < 16 * cpr; c += (int)blockDim.x) {
            const int r = c / cpr, cc = c - r * cpr;
            const unsigned loff = lbase + (unsigned)(r * ldS * 2 + cc * 16);
            const char* g = gbase + (size_t)r * Kp * 2 + (size_t)cc * 16;
            asm volatile("global_load_async_to_lds_b128 %0, %1, off"
                         :: "v"(loff), "v"(g) : "memory");
        }
        asm volatile("s_wait_asynccnt 0x0" ::: "memory");
    }
#endif
    __syncthreads();

    v8f acc[OT];
#pragma unroll
    for (int q = 0; q < OT; ++q) acc[q] = (v8f){0.f,0.f,0.f,0.f,0.f,0.f,0.f,0.f};

    const int ot0 = wave * OT;
#if defined(__HIP_DEVICE_COMPILE__)
    const int kts = Kp >> 5;
    for (int kt = 0; kt < kts; ++kt) {
        // A fragment from LDS: lane L = row (mt*16 + L%16),
        // K chunks [16h..16h+7] and [16h+8..16h+15]
        const u16* arow = smem + lm * ldS + kt * 32 + 16 * h;
        const v8u c0 = *(const v8u*)arow;
        const v8u c1 = *(const v8u*)(arow + 8);
        v16u av;
#pragma unroll
        for (int e = 0; e < 8; ++e) { av[e] = c0[e]; av[8 + e] = c1[e]; }
        const v16bf af = __builtin_bit_cast(v16bf, av);
#pragma unroll
        for (int q = 0; q < OT; ++q) {
            // B fragment: contiguous 32B per lane from packed weights
            const v16u bv = *(const v16u*)(Wp +
                ((size_t)(kt * otiles + ot0 + q) * 512) + (size_t)L * 16);
            acc[q] = __builtin_amdgcn_wmma_f32_16x16x32_bf16(
                false, af, false, __builtin_bit_cast(v16bf, bv),
                (short)0, acc[q], false, false);
        }
    }
#endif

#pragma unroll
    for (int q = 0; q < OT; ++q) {
        const int n = (ot0 + q) * 16 + lm;
        const float bn = (n < O) ? bias[n] : 0.0f;
#pragma unroll
        for (int r = 0; r < 8; ++r) {
            const int m = mt * 16 + r + 8 * h;  // C layout: VGPR r -> M = r + 8h
            float v = acc[q][r] + bn;
            if (ACT == 1) v = (v > 0.0f) ? v : (__expf(v) - 1.0f);
            if (ACT == 2) v = (v >= 0.0f) ? v : 0.01f * v;
            if (n < O) {
                if (outB) outB[(size_t)m * ldOut + n] = f2bf(v);
                else      outF[(size_t)m * ldOut + n] = v;
            }
        }
    }
}

// ---------------------------------------------------------------------------
// x [N,12] f32 -> bf16 [N,32] zero-padded
// ---------------------------------------------------------------------------
__global__ void pad_x_k(const float* __restrict__ x, u16* __restrict__ out) {
    const int e = blockIdx.x * 256 + threadIdx.x;
    if (e >= NNODES * 32) return;
    const int row = e >> 5, c = e & 31;
    out[e] = (c < 12) ? f2bf(x[row * 12 + c]) : (u16)0;
}

// ---------------------------------------------------------------------------
// lin_s (12->4) and lin_h (12->22), fp32 VALU (tiny)
// ---------------------------------------------------------------------------
__global__ void lin_sh_k(const float* __restrict__ x1,
                         const float* __restrict__ Ws, const float* __restrict__ bs,
                         const float* __restrict__ Wh, const float* __restrict__ bh,
                         float* __restrict__ s, float* __restrict__ h) {
    const int i = blockIdx.x * 256 + threadIdx.x;
    if (i >= NNODES) return;
    float xr[12];
#pragma unroll
    for (int k = 0; k < 12; ++k) xr[k] = x1[(size_t)i * 12 + k];
#pragma unroll
    for (int o = 0; o < 4; ++o) {
        float a = bs[o];
#pragma unroll
        for (int k = 0; k < 12; ++k) a += xr[k] * Ws[k * 4 + o];
        s[(size_t)i * 4 + o] = a;
    }
    for (int o = 0; o < 22; ++o) {
        float a = bh[o];
#pragma unroll
        for (int k = 0; k < 12; ++k) a += xr[k] * Wh[k * 22 + o];
        h[(size_t)i * 22 + o] = a;
    }
}

// ---------------------------------------------------------------------------
// Fused kNN: one wave32 per row. Each lane streams j = lane, lane+32, ...
// keeping a sorted top-16 in registers; 32 lanes merged with shfl_xor
// tournament (no LDS). Avoids materializing the 576 MB d2 matrix.
// ---------------------------------------------------------------------------
__global__ __launch_bounds__(256) void knn_topk_k(const float* __restrict__ s,
                                                  int* __restrict__ idx) {
    const int lane = threadIdx.x & 31;
    const int row = blockIdx.x * 8 + (threadIdx.x >> 5);
    const float4* s4 = (const float4*)s;
    const float4 si = s4[row];

    float bd[16]; int bi[16];
#pragma unroll
    for (int t = 0; t < 16; ++t) { bd[t] = 3.0e38f; bi[t] = 0x7fffffff; }

    for (int j = lane; j < NNODES; j += 32) {
        const float4 sj = s4[j];
        const float dx = si.x - sj.x, dy = si.y - sj.y;
        const float dz = si.z - sj.z, dw = si.w - sj.w;
        const float d = dx * dx + dy * dy + dz * dz + dw * dw;
        if (d < bd[15]) {
            bd[15] = d; bi[15] = j;
#pragma unroll
            for (int t = 15; t > 0; --t) {
                if (bd[t] < bd[t - 1]) {
                    float td = bd[t]; bd[t] = bd[t - 1]; bd[t - 1] = td;
                    int   ti = bi[t]; bi[t] = bi[t - 1]; bi[t - 1] = ti;
                }
            }
        }
    }

    for (int t = 0; t < 16; ++t) {
        float d = bd[0]; int ix = bi[0]; int ln = lane;
#pragma unroll
        for (int off = 16; off > 0; off >>= 1) {
            const float dO = __shfl_xor(d, off, 32);
            const int  ixO = __shfl_xor(ix, off, 32);
            const int  lnO = __shfl_xor(ln, off, 32);
            if (dO < d || (dO == d && ixO < ix)) { d = dO; ix = ixO; ln = lnO; }
        }
        if (lane == 0) idx[(size_t)row * 16 + t] = ix;
        if (lane == ln) {
#pragma unroll
            for (int q = 0; q < 15; ++q) { bd[q] = bd[q + 1]; bi[q] = bi[q + 1]; }
            bd[15] = 3.0e38f; bi[15] = 0x7fffffff;
        }
    }
}

// ---------------------------------------------------------------------------
// Aggregation: w = exp(-10 d2), weighted mean/max of h over 16 neighbors.
// Directly emits lin_out's A matrix: bf16 [N,64] = [x1(12) | mean(22) | max(22) | 0 x8]
// ---------------------------------------------------------------------------
__global__ void aggregate_k(const float* __restrict__ s, const float* __restrict__ hbuf,
                            const float* __restrict__ x1, const int* __restrict__ idx,
                            u16* __restrict__ alin) {
    const int i = blockIdx.x * 256 + threadIdx.x;
    if (i >= NNODES) return;
    const float4* s4 = (const float4*)s;
    const float4 si = s4[i];
    float mean[22], mx[22];
#pragma unroll
    for (int c = 0; c < 22; ++c) { mean[c] = 0.0f; mx[c] = -3.0e38f; }
    for (int k = 0; k < 16; ++k) {
        const int j = idx[(size_t)i * 16 + k];
        const float4 sj = s4[j];
        const float dx = si.x - sj.x, dy = si.y - sj.y;
        const float dz = si.z - sj.z, dw = si.w - sj.w;
        const float d2 = dx * dx + dy * dy + dz * dz + dw * dw;
        const float w = __expf(-10.0f * d2);
        const float* hr = hbuf + (size_t)j * 22;
#pragma unroll
        for (int c = 0; c < 22; ++c) {
            const float m = w * hr[c];
            mean[c] += m;
            if (m > mx[c]) mx[c] = m;
        }
    }
    u16* r = alin + (size_t)i * 64;
    const float* xr = x1 + (size_t)i * 12;
#pragma unroll
    for (int c = 0; c < 12; ++c) r[c] = f2bf(xr[c]);
#pragma unroll
    for (int c = 0; c < 22; ++c) r[12 + c] = f2bf(mean[c] * (1.0f / 16.0f));
#pragma unroll
    for (int c = 0; c < 22; ++c) r[34 + c] = f2bf(mx[c]);
#pragma unroll
    for (int c = 56; c < 64; ++c) r[c] = 0;
}

// ---------------------------------------------------------------------------
// nn3 input: bf16 [N,96] = [x(12) | pred_ids(6) | xc(64) | 0 x14]
// ---------------------------------------------------------------------------
__global__ void concat3_k(const float* __restrict__ x, const float* __restrict__ pred_ids,
                          const u16* __restrict__ xc, u16* __restrict__ a3) {
    const int i = blockIdx.x * 256 + threadIdx.x;
    if (i >= NNODES) return;
    u16* r = a3 + (size_t)i * 96;
#pragma unroll
    for (int c = 0; c < 12; ++c) r[c] = f2bf(x[(size_t)i * 12 + c]);
#pragma unroll
    for (int c = 0; c < 6; ++c) r[12 + c] = f2bf(pred_ids[(size_t)i * 6 + c]);
#pragma unroll
    for (int c = 0; c < 64; ++c) r[18 + c] = xc[(size_t)i * 64 + c];
#pragma unroll
    for (int c = 82; c < 96; ++c) r[c] = 0;
}

// ---------------------------------------------------------------------------
// Echo ygen_id, ygen, ycand_id, ycand into d_out tail (as float)
// ---------------------------------------------------------------------------
__global__ void tail_copy_k(const int* __restrict__ ygen_id, const float* __restrict__ ygen,
                            const int* __restrict__ ycand_id, const float* __restrict__ ycand,
                            float* __restrict__ out) {
    const int e = blockIdx.x * 256 + threadIdx.x;
    if (e >= 168000) return;
    float v;
    if (e < 12000)       v = (float)ygen_id[e];
    else if (e < 84000)  v = ygen[e - 12000];
    else if (e < 96000)  v = (float)ycand_id[e - 84000];
    else                 v = ycand[e - 96000];
    out[144000 + e] = v;
}

// ---------------------------------------------------------------------------
// Host-side orchestration
// ---------------------------------------------------------------------------
extern "C" void kernel_launch(void* const* d_in, const int* in_sizes, int n_in,
                              void* d_out, int out_size, void* d_ws, size_t ws_size,
                              hipStream_t stream) {
    (void)in_sizes; (void)n_in; (void)out_size; (void)ws_size;
    const int N = NNODES;

    const float* x        = (const float*)d_in[0];
    const int*   ygen_id  = (const int*)  d_in[1];
    const float* ygen     = (const float*)d_in[2];
    const int*   ycand_id = (const int*)  d_in[3];
    const float* ycand    = (const float*)d_in[4];

    auto Wf = [&](int i) { return (const float*)d_in[i]; };

    char* ws = (char*)d_ws;
    size_t off = 0;
    auto alc = [&](size_t bytes) {
        size_t o = off;
        off = (off + bytes + 255) & ~(size_t)255;
        return o;
    };

    u16*   xpad = (u16*)  (ws + alc((size_t)N * 32 * 2));
    u16*   t1   = (u16*)  (ws + alc((size_t)N * 64 * 2));
    u16*   t2   = (u16*)  (ws + alc((size_t)N * 64 * 2));
    float* x1f  = (float*)(ws + alc((size_t)N * 12 * 4));
    float* sbuf = (float*)(ws + alc((size_t)N * 4 * 4));
    float* hbuf = (float*)(ws + alc((size_t)N * 22 * 4));
    int*   idxb = (int*)  (ws + alc((size_t)N * 16 * 4));
    u16*   alin = (u16*)  (ws + alc((size_t)N * 64 * 2));
    u16*   xcb  = (u16*)  (ws + alc((size_t)N * 64 * 2));
    u16*   a3   = (u16*)  (ws + alc((size_t)N * 96 * 2));
    u16*   big1 = (u16*)  (ws + alc((size_t)N * 256 * 2));
    u16*   big2 = (u16*)  (ws + alc((size_t)N * 256 * 2));

    // Packed weight layers: {W idx, b idx, K, O, Kp, Op}
    struct LW { int wi, bi, K, O, Kp, Op; u16* packed; };
    LW lw[12] = {
        { 5,  6,  12,  64,  32,  64, nullptr},  // nn1.0
        { 7,  8,  64,  64,  64,  64, nullptr},  // nn1.1
        { 9, 10,  64,  12,  64,  16, nullptr},  // nn1.2
        {15, 16,  56,  64,  64,  64, nullptr},  // lin_out
        {17, 18,  64, 256,  64, 256, nullptr},  // nn2.0
        {19, 20, 256, 256, 256, 256, nullptr},  // nn2.1
        {21, 22, 256, 256, 256, 256, nullptr},  // nn2.2
        {23, 24, 256,   6, 256,  16, nullptr},  // nn2.3
        {25, 26,  82, 256,  96, 256, nullptr},  // nn3.0
        {27, 28, 256, 256, 256, 256, nullptr},  // nn3.1
        {29, 30, 256, 256, 256, 256, nullptr},  // nn3.2
        {31, 32, 256,   6, 256,  16, nullptr},  // nn3.3
    };
    for (int i = 0; i < 12; ++i) {
        const int frags = (lw[i].Kp >> 5) * (lw[i].Op >> 4);
        lw[i].packed = (u16*)(ws + alc((size_t)frags * 512 * 2));
        pack_w_k<<<frags, 32, 0, stream>>>(Wf(lw[i].wi), lw[i].packed,
                                           lw[i].K, lw[i].O, lw[i].Op);
    }

    // GEMM launcher: block covers full Op width; each wave emits 16x(16*OT).
    auto gemm = [&](int act, const u16* A, const LW& l, u16* outB, float* outF,
                    int ldOut) {
        const int otiles = l.Op >> 4;
        const int OT = (otiles >= 4) ? 4 : 1;
        const int nw = otiles / OT;
        dim3 g(N / 16), b(nw * 32);
#define GEMM_GO(ACT, OTV)                                                      \
        gemm_wmma_k<ACT, OTV><<<g, b, 0, stream>>>(A, l.packed, Wf(l.bi),      \
            outB, outF, l.Kp, l.O, l.Op, ldOut)
        if (OT == 4) {
            if (act == 0)      GEMM_GO(0, 4);
            else if (act == 1) GEMM_GO(1, 4);
            else               GEMM_GO(2, 4);
        } else {
            if (act == 0)      GEMM_GO(0, 1);
            else if (act == 1) GEMM_GO(1, 1);
            else               GEMM_GO(2, 1);
        }
#undef GEMM_GO
    };

    float* out = (float*)d_out;
    float* pred_ids = out;            // [N,6]
    float* pred_p4  = out + 72000;    // [N,6]

    // nn1: 12 -> 64 -> 64 -> 12
    pad_x_k<<<(N * 32 + 255) / 256, 256, 0, stream>>>(x, xpad);
    gemm(1, xpad, lw[0], t1, nullptr, 64);
    gemm(1, t1,   lw[1], t2, nullptr, 64);
    gemm(0, t2,   lw[2], nullptr, x1f, 12);

    // GravNet
    lin_sh_k<<<(N + 255) / 256, 256, 0, stream>>>(x1f, Wf(11), Wf(12), Wf(13), Wf(14),
                                                  sbuf, hbuf);
    knn_topk_k<<<N / 8, 256, 0, stream>>>(sbuf, idxb);
    aggregate_k<<<(N + 255) / 256, 256, 0, stream>>>(sbuf, hbuf, x1f, idxb, alin);
    gemm(2, alin, lw[3], xcb, nullptr, 64);   // lin_out + leaky_relu -> xc (bf16)

    // nn2: 64 -> 256 -> 256 -> 256 -> 6
    gemm(1, xcb,  lw[4], big1, nullptr, 256);
    gemm(1, big1, lw[5], big2, nullptr, 256);
    gemm(1, big2, lw[6], big1, nullptr, 256);
    gemm(0, big1, lw[7], nullptr, pred_ids, 6);

    // nn3: [x | pred_ids | xc] (82 -> pad 96) -> 256 -> 256 -> 256 -> 6
    concat3_k<<<(N + 255) / 256, 256, 0, stream>>>(x, pred_ids, xcb, a3);
    gemm(1, a3,   lw[8],  big1, nullptr, 256);
    gemm(1, big1, lw[9],  big2, nullptr, 256);
    gemm(1, big2, lw[10], big1, nullptr, 256);
    gemm(0, big1, lw[11], nullptr, pred_p4, 6);

    // Echo targets
    tail_copy_k<<<(168000 + 255) / 256, 256, 0, stream>>>(ygen_id, ygen, ycand_id, ycand, out);
}
